// MGNProcessor_37821482008638
// MI455X (gfx1250) — compile-verified, matrix operand order checked by reading, not run on hardware
//
#include <hip/hip_runtime.h>

// ---------------------------------------------------------------------------
// MeshGraphNet-style processor for MI455X (gfx1250), wave32 + WMMA bf16
// + double-buffered LDS weight panels fed by async global->LDS copies.
// ---------------------------------------------------------------------------

#define NN 50000
#define EE 600000
#define LL 128

typedef __attribute__((ext_vector_type(16))) __bf16 v16bf;
typedef __attribute__((ext_vector_type(8)))  float  v8f;
typedef __attribute__((ext_vector_type(4)))  int    v4i;

#define WMMA_BF16(A, B, C) \
  __builtin_amdgcn_wmma_f32_16x16x32_bf16(false, (A), false, (B), (short)0, (C), false, false)

#if __has_builtin(__builtin_amdgcn_global_load_async_to_lds_b128)
#define HAVE_ASYNC_LDS 1
typedef __attribute__((address_space(1))) v4i as1_v4i;   // global int4
typedef __attribute__((address_space(3))) v4i as3_v4i;   // LDS int4
#else
#define HAVE_ASYNC_LDS 0
#endif

// Issue one 8KB weight panel (kc) into LDS buffer buf (256 threads x 2 x 16B).
__device__ __forceinline__ void panel_issue(const __bf16* __restrict__ wg,
                                            __bf16* sh, int kc, int buf, int tid) {
  const char* g = (const char*)wg + (size_t)kc * 8192;
  char* l = (char*)sh + buf * 8192;
#if HAVE_ASYNC_LDS
#pragma unroll
  for (int i = 0; i < 2; ++i) {
    int off = (i * 256 + tid) * 16;
    __builtin_amdgcn_global_load_async_to_lds_b128((as1_v4i*)(g + off),
                                                   (as3_v4i*)(l + off), 0, 0);
  }
#else
#pragma unroll
  for (int i = 0; i < 2; ++i) {
    int idx = i * 256 + tid;
    ((uint4*)l)[idx] = ((const uint4*)g)[idx];
  }
#endif
}

__device__ __forceinline__ void panel_wait() {
#if HAVE_ASYNC_LDS
#if __has_builtin(__builtin_amdgcn_s_wait_asynccnt)
  __builtin_amdgcn_s_wait_asynccnt(0);
#else
  asm volatile("s_wait_asynccnt 0x0" ::: "memory");
#endif
#endif
}

// Double-buffered K-loop: panel kc+1 streams into LDS while panel kc feeds
// 8 WMMAs per wave from ds loads. afn(kc) builds the A fragment.
template <int KC, typename AF>
__device__ __forceinline__ void wmma_pipe(AF afn, const __bf16* __restrict__ wg,
                                          __bf16* sh, v8f* acc, int tid, bool act) {
  const int lane = tid & 31;
  panel_issue(wg, sh, 0, 0, tid);
#pragma unroll
  for (int kc = 0; kc < KC; ++kc) {
    panel_wait();
    __syncthreads();                       // panel kc resident; prev reads done
    if (kc + 1 < KC) panel_issue(wg, sh, kc + 1, (kc + 1) & 1, tid);
    if (act) {
      v16bf a = afn(kc);
      const __bf16* bp = sh + (kc & 1) * 4096;
#pragma unroll
      for (int nt = 0; nt < 8; ++nt) {
        v16bf b = *(const v16bf*)(bp + (nt * 32 + lane) * 16);
        acc[nt] = WMMA_BF16(a, b, acc[nt]);
      }
    }
  }
}

// A-fragment k-index pattern for 16-bit 16x32 A tiles (ISA 7.12.2):
//   k = (v>>2)*16 + (lane>>4)*8 + (v&3)*2 + p ; row m = lane & 15
__device__ __forceinline__ v16bf load_a_row(const float* __restrict__ rp, int kk0, int lane) {
  v16bf a;
#pragma unroll
  for (int v = 0; v < 8; ++v) {
    int klo = kk0 + ((v >> 2) << 4) + ((lane >> 4) << 3) + ((v & 3) << 1);
    float2 f = *(const float2*)(rp + klo);
    a[2 * v]     = (__bf16)f.x;
    a[2 * v + 1] = (__bf16)f.y;
  }
  return a;
}

__device__ __forceinline__ v16bf load_a_bnrelu(const float* __restrict__ zrow,
                                               const float* __restrict__ affa,
                                               const float* __restrict__ affb,
                                               int kk0, int lane) {
  v16bf a;
#pragma unroll
  for (int v = 0; v < 8; ++v) {
    int klo = kk0 + ((v >> 2) << 4) + ((lane >> 4) << 3) + ((v & 3) << 1);
    float2 zf = *(const float2*)(zrow + klo);
    float2 av = *(const float2*)(affa + klo);
    float2 bv = *(const float2*)(affb + klo);
    a[2 * v]     = (__bf16)fmaxf(fmaf(zf.x, av.x, bv.x), 0.f);
    a[2 * v + 1] = (__bf16)fmaxf(fmaf(zf.y, av.y, bv.y), 0.f);
  }
  return a;
}

// ---------------------------------------------------------------------------
// Weight conversion: fp32 [K][128] row-major -> bf16 fragments in WMMA order.
// ---------------------------------------------------------------------------
__global__ void k_cvt_wb(const float* __restrict__ src, __bf16* __restrict__ dst, int K) {
  int t = blockIdx.x * blockDim.x + threadIdx.x;
  if (t >= K * LL) return;
  int e    = t & 15;
  int lane = (t >> 4) & 31;
  int tl   = t >> 9;           // kc*8 + nt
  int kc   = tl >> 3;
  int nt   = tl & 7;
  int v = e >> 1, p = e & 1;
  int k = kc * 32 + ((v >> 2) << 4) + ((lane >> 4) << 3) + ((v & 3) << 1) + p;
  int n = nt * 16 + (lane & 15);
  dst[t] = (__bf16)src[(size_t)k * LL + n];
}

__global__ void k_zero(float* __restrict__ p, size_t n) {
  size_t i = (size_t)blockIdx.x * blockDim.x + threadIdx.x;
  if (i < n) p[i] = 0.f;
}

__global__ void k_copy4(const float4* __restrict__ s, float4* __restrict__ d, size_t n) {
  size_t i = (size_t)blockIdx.x * blockDim.x + threadIdx.x;
  if (i < n) d[i] = s[i];
}

__global__ void k_finalize(const float* __restrict__ ssum, const float* __restrict__ sssq,
                           const float* __restrict__ g, const float* __restrict__ be,
                           float rinv, float* __restrict__ affa, float* __restrict__ affb) {
  int c = threadIdx.x;
  float mean = ssum[c] * rinv;
  float var  = sssq[c] * rinv - mean * mean;
  float a    = g[c] * __frsqrt_rn(var + 1e-5f);
  affa[c] = a;
  affb[c] = be[c] - mean * a;
}

// ---------------------------------------------------------------------------
// Edge MLP layer 1: z = concat(x[dst], x[src], e) @ W1 + b1, plus BN stats.
// One wave per 16-edge x 128-channel tile; K = 384 (12 bf16 chunks).
// ---------------------------------------------------------------------------
__global__ void __launch_bounds__(256)
k_edge_mlp1(const float* __restrict__ x, const float* __restrict__ e,
            const long long* __restrict__ eidx, const __bf16* __restrict__ w1,
            const float* __restrict__ b1, float* __restrict__ z,
            float* __restrict__ ssum, float* __restrict__ sssq) {
  __shared__ __bf16 bsh[2 * 4096];
  __shared__ float sm_sum[LL];
  __shared__ float sm_ssq[LL];
  const int tid = threadIdx.x;
  if (tid < LL) { sm_sum[tid] = 0.f; sm_ssq[tid] = 0.f; }

  const int lane = tid & 31;
  const int tile = blockIdx.x * 8 + (tid >> 5);
  const bool act = tile < (EE / 16);

  const float* rp0 = x;
  const float* rp1 = x;
  const float* rp2 = e;
  if (act) {
    const int m = lane & 15;
    const size_t erow = (size_t)tile * 16 + m;
    const long long srcI = eidx[erow];        // edge_index[0] = source
    const long long dstI = eidx[EE + erow];   // edge_index[1] = target
    rp0 = x + (size_t)dstI * LL;              // concat: [x[dst], x[src], e]
    rp1 = x + (size_t)srcI * LL;
    rp2 = e + erow * LL;
    __builtin_prefetch(rp1, 0, 0);            // segments used from kc>=4
    __builtin_prefetch(rp1 + 64, 0, 0);
    __builtin_prefetch(rp2, 0, 0);
    __builtin_prefetch(rp2 + 64, 0, 0);
  }

  const v8f zv8 = {0.f, 0.f, 0.f, 0.f, 0.f, 0.f, 0.f, 0.f};
  v8f acc[8];
#pragma unroll
  for (int nt = 0; nt < 8; ++nt) acc[nt] = zv8;

  wmma_pipe<12>(
      [&](int kc) {
        const float* rp = (kc < 4) ? rp0 : (kc < 8) ? rp1 : rp2;
        return load_a_row(rp, (kc & 3) * 32, lane);
      },
      w1, bsh, acc, tid, act);

  if (act) {
    const int half = lane >> 4;
#pragma unroll
    for (int nt = 0; nt < 8; ++nt) {
      const int c = nt * 16 + (lane & 15);
      const float bv = b1[c];
      float ps = 0.f, pq = 0.f;
#pragma unroll
      for (int r = 0; r < 8; ++r) {
        float zval = acc[nt][r] + bv;
        ps += zval;
        pq += zval * zval;
        z[((size_t)tile * 16 + r + half * 8) * LL + c] = zval;
      }
      ps += __shfl_xor(ps, 16, 32);
      pq += __shfl_xor(pq, 16, 32);
      if (lane < 16) {
        atomicAdd(&sm_sum[c], ps);
        atomicAdd(&sm_ssq[c], pq);
      }
    }
  }
  __syncthreads();
  if (tid < LL) {
    atomicAdd(&ssum[tid], sm_sum[tid]);
    atomicAdd(&sssq[tid], sm_ssq[tid]);
  }
}

// ---------------------------------------------------------------------------
// Edge MLP layer 2: h = relu(BN(z)) @ W2 + b2; e' = relu(h)+e; agg += e'.
// ---------------------------------------------------------------------------
__global__ void __launch_bounds__(256)
k_edge_mlp2(const float* __restrict__ z, const float* __restrict__ affa,
            const float* __restrict__ affb, const __bf16* __restrict__ w2,
            const float* __restrict__ b2, const long long* __restrict__ eidx,
            float* __restrict__ eo, float* __restrict__ agg) {
  __shared__ __bf16 bsh[2 * 4096];
  const int tid  = threadIdx.x;
  const int lane = tid & 31;
  const int tile = blockIdx.x * 8 + (tid >> 5);
  const bool act = tile < (EE / 16);
  const int m = lane & 15;
  const float* zrow = z + ((size_t)(act ? tile : 0) * 16 + m) * LL;

  const v8f zv8 = {0.f, 0.f, 0.f, 0.f, 0.f, 0.f, 0.f, 0.f};
  v8f acc[8];
#pragma unroll
  for (int nt = 0; nt < 8; ++nt) acc[nt] = zv8;

  wmma_pipe<4>(
      [&](int kc) { return load_a_bnrelu(zrow, affa, affb, kc * 32, lane); },
      w2, bsh, acc, tid, act);

  if (act) {
    const int half = lane >> 4;
    long long drow[8];
#pragma unroll
    for (int r = 0; r < 8; ++r)
      drow[r] = eidx[EE + (size_t)tile * 16 + r + half * 8];

#pragma unroll
    for (int nt = 0; nt < 8; ++nt) {
      const int c = nt * 16 + (lane & 15);
      const float bv = b2[c];
#pragma unroll
      for (int r = 0; r < 8; ++r) {
        size_t row = (size_t)tile * 16 + r + half * 8;
        float h = fmaxf(acc[nt][r] + bv, 0.f);
        float res = h + eo[row * LL + c];   // residual: new edge_attr
        eo[row * LL + c] = res;
        atomicAdd(&agg[(size_t)drow[r] * LL + c], res); // segment-sum
      }
    }
  }
}

// ---------------------------------------------------------------------------
// Node MLP layer 1: z = concat(x, agg) @ W1 + b1, plus BN stats. K = 256.
// ---------------------------------------------------------------------------
__global__ void __launch_bounds__(256)
k_node_mlp1(const float* __restrict__ x, const float* __restrict__ agg,
            const __bf16* __restrict__ w1, const float* __restrict__ b1,
            float* __restrict__ z, float* __restrict__ ssum, float* __restrict__ sssq) {
  __shared__ __bf16 bsh[2 * 4096];
  __shared__ float sm_sum[LL];
  __shared__ float sm_ssq[LL];
  const int tid = threadIdx.x;
  if (tid < LL) { sm_sum[tid] = 0.f; sm_ssq[tid] = 0.f; }

  const int lane = tid & 31;
  const int tile = blockIdx.x * 8 + (tid >> 5);
  const bool act = tile < (NN / 16);
  const size_t nrow = (size_t)(act ? tile : 0) * 16 + (lane & 15);
  const float* rp0 = x + nrow * LL;
  const float* rp1 = agg + nrow * LL;

  const v8f zv8 = {0.f, 0.f, 0.f, 0.f, 0.f, 0.f, 0.f, 0.f};
  v8f acc[8];
#pragma unroll
  for (int nt = 0; nt < 8; ++nt) acc[nt] = zv8;

  wmma_pipe<8>(
      [&](int kc) {
        const float* rp = (kc < 4) ? rp0 : rp1;
        return load_a_row(rp, (kc & 3) * 32, lane);
      },
      w1, bsh, acc, tid, act);

  if (act) {
    const int half = lane >> 4;
#pragma unroll
    for (int nt = 0; nt < 8; ++nt) {
      const int c = nt * 16 + (lane & 15);
      const float bv = b1[c];
      float ps = 0.f, pq = 0.f;
#pragma unroll
      for (int r = 0; r < 8; ++r) {
        float zval = acc[nt][r] + bv;
        ps += zval;
        pq += zval * zval;
        z[((size_t)tile * 16 + r + half * 8) * LL + c] = zval;
      }
      ps += __shfl_xor(ps, 16, 32);
      pq += __shfl_xor(pq, 16, 32);
      if (lane < 16) {
        atomicAdd(&sm_sum[c], ps);
        atomicAdd(&sm_ssq[c], pq);
      }
    }
  }
  __syncthreads();
  if (tid < LL) {
    atomicAdd(&ssum[tid], sm_sum[tid]);
    atomicAdd(&sssq[tid], sm_ssq[tid]);
  }
}

// ---------------------------------------------------------------------------
// Node MLP layer 2: x' = relu(relu(BN(z)) @ W2 + b2) + x.
// ---------------------------------------------------------------------------
__global__ void __launch_bounds__(256)
k_node_mlp2(const float* __restrict__ z, const float* __restrict__ affa,
            const float* __restrict__ affb, const __bf16* __restrict__ w2,
            const float* __restrict__ b2, float* __restrict__ xo) {
  __shared__ __bf16 bsh[2 * 4096];
  const int tid  = threadIdx.x;
  const int lane = tid & 31;
  const int tile = blockIdx.x * 8 + (tid >> 5);
  const bool act = tile < (NN / 16);
  const int m = lane & 15;
  const float* zrow = z + ((size_t)(act ? tile : 0) * 16 + m) * LL;

  const v8f zv8 = {0.f, 0.f, 0.f, 0.f, 0.f, 0.f, 0.f, 0.f};
  v8f acc[8];
#pragma unroll
  for (int nt = 0; nt < 8; ++nt) acc[nt] = zv8;

  wmma_pipe<4>(
      [&](int kc) { return load_a_bnrelu(zrow, affa, affb, kc * 32, lane); },
      w2, bsh, acc, tid, act);

  if (act) {
    const int half = lane >> 4;
#pragma unroll
    for (int nt = 0; nt < 8; ++nt) {
      const int c = nt * 16 + (lane & 15);
      const float bv = b2[c];
#pragma unroll
      for (int r = 0; r < 8; ++r) {
        size_t row = (size_t)tile * 16 + r + half * 8;
        float h = fmaxf(acc[nt][r] + bv, 0.f);
        xo[row * LL + c] = h + xo[row * LL + c];  // residual
      }
    }
  }
}

// ---------------------------------------------------------------------------
// Host-side orchestration.
// ---------------------------------------------------------------------------
extern "C" void kernel_launch(void* const* d_in, const int* in_sizes, int n_in,
                              void* d_out, int out_size, void* d_ws, size_t ws_size,
                              hipStream_t stream) {
  (void)in_sizes; (void)n_in; (void)out_size; (void)ws_size;

  const float*     x    = (const float*)d_in[0];
  const float*     ea   = (const float*)d_in[1];
  const long long* eidx = (const long long*)d_in[2];   // int64 [2, E]
  const float* eW1  = (const float*)d_in[3];
  const float* eb1  = (const float*)d_in[4];
  const float* eg1  = (const float*)d_in[5];
  const float* ebe1 = (const float*)d_in[6];
  const float* eW2  = (const float*)d_in[7];
  const float* eb2  = (const float*)d_in[8];
  const float* nW1  = (const float*)d_in[9];
  const float* nb1  = (const float*)d_in[10];
  const float* ng1  = (const float*)d_in[11];
  const float* nbe1 = (const float*)d_in[12];
  const float* nW2  = (const float*)d_in[13];
  const float* nb2  = (const float*)d_in[14];

  // d_out holds the evolving state: x [N,L] then edge_attr [E,L].
  float* xo = (float*)d_out;
  float* eo = xo + (size_t)NN * LL;

  // Workspace layout.
  __bf16* we1 = (__bf16*)d_ws;                       // 4 x 384*128
  __bf16* we2 = we1 + (size_t)4 * 384 * LL;          // 4 x 128*128
  __bf16* wn1 = we2 + (size_t)4 * 128 * LL;          // 4 x 256*128
  __bf16* wn2 = wn1 + (size_t)4 * 256 * LL;          // 4 x 128*128
  float* fb    = (float*)(wn2 + (size_t)4 * 128 * LL);
  float* ssumE = fb;                 // 128
  float* sssqE = fb + 128;           // 128
  float* ssumN = fb + 256;           // 128
  float* sssqN = fb + 384;           // 128
  float* agg   = fb + 512;           // N*L  (stats+agg zeroed together)
  float* affa  = agg + (size_t)NN * LL;    // 128
  float* affb  = affa + 128;               // 128
  float* ze    = affb + 128;               // E*L pre-BN edge activations
  float* zn    = ze + (size_t)EE * LL;     // N*L pre-BN node activations

  // One-time: weights -> bf16 WMMA-fragment order.
  for (int s = 0; s < 4; ++s) {
    k_cvt_wb<<<(384 * LL + 255) / 256, 256, 0, stream>>>(eW1 + (size_t)s * 384 * LL, we1 + (size_t)s * 384 * LL, 384);
    k_cvt_wb<<<(128 * LL + 255) / 256, 256, 0, stream>>>(eW2 + (size_t)s * 128 * LL, we2 + (size_t)s * 128 * LL, 128);
    k_cvt_wb<<<(256 * LL + 255) / 256, 256, 0, stream>>>(nW1 + (size_t)s * 256 * LL, wn1 + (size_t)s * 256 * LL, 256);
    k_cvt_wb<<<(128 * LL + 255) / 256, 256, 0, stream>>>(nW2 + (size_t)s * 128 * LL, wn2 + (size_t)s * 128 * LL, 128);
  }

  // Initialize evolving state from inputs (inputs never mutated).
  {
    size_t nx = (size_t)NN * LL / 4, ne = (size_t)EE * LL / 4;
    k_copy4<<<(unsigned)((nx + 255) / 256), 256, 0, stream>>>((const float4*)x, (float4*)xo, nx);
    k_copy4<<<(unsigned)((ne + 255) / 256), 256, 0, stream>>>((const float4*)ea, (float4*)eo, ne);
  }

  const int EBLK = (EE / 16 + 7) / 8;   // 4688 blocks of 8 waves
  const int NBLK = (NN / 16 + 7) / 8;   // 391 blocks of 8 waves

  for (int s = 0; s < 4; ++s) {
    size_t nz = 512 + (size_t)NN * LL;  // stats (E+N) + agg
    k_zero<<<(unsigned)((nz + 255) / 256), 256, 0, stream>>>(fb, nz);

    k_edge_mlp1<<<EBLK, 256, 0, stream>>>(xo, eo, eidx, we1 + (size_t)s * 384 * LL,
                                          eb1 + s * LL, ze, ssumE, sssqE);
    k_finalize<<<1, 128, 0, stream>>>(ssumE, sssqE, eg1 + s * LL, ebe1 + s * LL,
                                      1.f / (float)EE, affa, affb);
    k_edge_mlp2<<<EBLK, 256, 0, stream>>>(ze, affa, affb, we2 + (size_t)s * 128 * LL,
                                          eb2 + s * LL, eidx, eo, agg);

    k_node_mlp1<<<NBLK, 256, 0, stream>>>(xo, agg, wn1 + (size_t)s * 256 * LL,
                                          nb1 + s * LL, zn, ssumN, sssqN);
    k_finalize<<<1, 128, 0, stream>>>(ssumN, sssqN, ng1 + s * LL, nbe1 + s * LL,
                                      1.f / (float)NN, affa, affb);
    k_node_mlp2<<<NBLK, 256, 0, stream>>>(zn, affa, affb, wn2 + (size_t)s * 128 * LL,
                                          nb2 + s * LL, xo);
  }
}
